// Attention_4183298146361
// MI455X (gfx1250) — matrix operand order repaired and checked by reading, not compiled, not run on hardware
//
#include <hip/hip_runtime.h>
#include <hip/hip_bf16.h>

typedef __attribute__((ext_vector_type(16))) _Float16 v16h;
typedef __attribute__((ext_vector_type(8)))  _Float16 v8h;
typedef __attribute__((ext_vector_type(8)))  float    v8f;
typedef __attribute__((ext_vector_type(4)))  unsigned int u32x4;
typedef __attribute__((ext_vector_type(8)))  int      i32x8;
typedef __attribute__((ext_vector_type(4)))  int      i32x4;

#define S_LEN   2048
#define IN_DIM  96
#define HIDDEN  256
#define NBATCH  8
#define NCHUNK  (S_LEN / 32)
#define NEGBIG  (-1.0e10f)
#define SCALE   0.0625f   /* 1/sqrt(256) */

__device__ __forceinline__ v8f wmma_f16(v16h a, v16h b, v8f c) {
  // D = A(16x32 f16) * B(32x16 f16) + C(16x16 f32)
  return __builtin_amdgcn_wmma_f32_16x16x32_f16(
      /*neg_a=*/false, a, /*neg_b=*/false, b,
      /*c_mod=*/(short)0, c, /*reuse_a=*/false, /*reuse_b=*/false);
}

__device__ __forceinline__ float rmax16(float v) {
  v = fmaxf(v, __shfl_xor(v, 8, 16));
  v = fmaxf(v, __shfl_xor(v, 4, 16));
  v = fmaxf(v, __shfl_xor(v, 2, 16));
  v = fmaxf(v, __shfl_xor(v, 1, 16));
  return v;
}
__device__ __forceinline__ float rsum16(float v) {
  v += __shfl_xor(v, 8, 16);
  v += __shfl_xor(v, 4, 16);
  v += __shfl_xor(v, 2, 16);
  v += __shfl_xor(v, 1, 16);
  return v;
}

// Build TDM D# group0: count=1 (valid), lds_addr, 57-bit global addr, type=2.
__device__ __forceinline__ u32x4 tdm_g0(unsigned lds_off, unsigned long long ga) {
  u32x4 g;
  g[0] = 1u;                                                  // count=1, user
  g[1] = lds_off;                                             // lds_addr
  g[2] = (unsigned)(ga & 0xFFFFFFFFull);                      // global[31:0]
  g[3] = (unsigned)((ga >> 32) & 0x01FFFFFFull) | (2u << 30); // global[56:32] | type=2
  return g;
}

// Build TDM D# group1 for a 2-D f16 tile: data_size=2B, no multicast/pad.
__device__ __forceinline__ i32x8 tdm_g1_2d(unsigned td0, unsigned td1,
                                           unsigned tile0, unsigned tile1,
                                           unsigned long long stride0) {
  i32x8 g;
  g[0] = 0x10000;                                    // data_size=1 (2 bytes)
  g[1] = (int)((td0 & 0xFFFFu) << 16);               // tensor_dim0[15:0] @63:48
  g[2] = (int)((td0 >> 16) | ((td1 & 0xFFFFu) << 16));
  g[3] = (int)((td1 >> 16) | (tile0 << 16));         // tile_dim0 @127:112
  g[4] = (int)tile1;                                 // tile_dim1; tile_dim2=0
  g[5] = (int)(stride0 & 0xFFFFFFFFull);             // tensor_dim0_stride[31:0]
  g[6] = (int)((stride0 >> 32) & 0xFFFFull);         // stride[47:32]; dim1_stride=0
  g[7] = 0;
  return g;
}

// ---------------------------------------------------------------------------
// Kernel 1: fused QKV projection.  X[16384,96](f32) @ W[96,256](f32) + b.
// Emits f16: Qh/Kh row-major [B*S, 256], Vt transposed [B, 256, S].
// grid = (1024, 4, 3), block = 128 (4 waves); one wave = one 16x16 tile.
// ---------------------------------------------------------------------------
__global__ __launch_bounds__(128) void proj_kernel(
    const float* __restrict__ q, const float* __restrict__ k,
    const float* __restrict__ v,
    const float* __restrict__ Wq, const float* __restrict__ Wk,
    const float* __restrict__ Wv,
    const float* __restrict__ bq, const float* __restrict__ bk,
    const float* __restrict__ bv,
    _Float16* __restrict__ Qh, _Float16* __restrict__ Kh,
    _Float16* __restrict__ Vt) {
  const int lane  = threadIdx.x & 31;
  const int w     = threadIdx.x >> 5;
  const int half_ = lane >> 4;
  const int l16   = lane & 15;
  const int row0  = blockIdx.x * 16;            // global row in [0,16384)
  const int n0    = (blockIdx.y * 4 + w) * 16;  // output column tile
  const int z     = blockIdx.z;                 // 0=Q 1=K 2=V

  const float* X    = (z == 0) ? q  : (z == 1) ? k  : v;
  const float* W    = (z == 0) ? Wq : (z == 1) ? Wk : Wv;
  const float* bias = (z == 0) ? bq : (z == 1) ? bk : bv;

  // A fragments: 3 chunks of K=32 (IN_DIM = 96)
  v16h a[3];
#pragma unroll
  for (int c = 0; c < 3; ++c) {
    const float* p = X + (size_t)(row0 + l16) * IN_DIM + 32 * c + 16 * half_;
    v8f lo = *(const v8f*)p;
    v8f hi = *(const v8f*)(p + 8);
#pragma unroll
    for (int i = 0; i < 8; ++i) {
      a[c][i]     = (_Float16)lo[i];
      a[c][i + 8] = (_Float16)hi[i];
    }
  }

  v8f acc = {0.f, 0.f, 0.f, 0.f, 0.f, 0.f, 0.f, 0.f};
#pragma unroll
  for (int c = 0; c < 3; ++c) {
    v16h bfr;
#pragma unroll
    for (int j = 0; j < 16; ++j)
      bfr[j] = (_Float16)W[(size_t)(32 * c + 16 * half_ + j) * HIDDEN + n0 + l16];
    acc = wmma_f16(a[c], bfr, acc);
  }

  const float bval = bias[n0 + l16];
#pragma unroll
  for (int r = 0; r < 8; ++r) acc[r] += bval;

  if (z < 2) {
    _Float16* out = (z == 0) ? Qh : Kh;
#pragma unroll
    for (int r = 0; r < 8; ++r) {
      const int grow = row0 + r + 8 * half_;
      out[(size_t)grow * HIDDEN + n0 + l16] = (_Float16)acc[r];
    }
  } else {  // V transposed: [B, HIDDEN, S]
#pragma unroll
    for (int r = 0; r < 8; ++r) {
      const int grow = row0 + r + 8 * half_;
      const int bb   = grow >> 11;
      const int ss   = grow & (S_LEN - 1);
      Vt[((size_t)bb * HIDDEN + (n0 + l16)) * S_LEN + ss] = (_Float16)acc[r];
    }
  }
}

// ---------------------------------------------------------------------------
// Kernel 2: flash attention + LayerNorm.  4 waves per block share one batch
// and consecutive q tiles; K/V tiles are staged into double-buffered LDS by
// the Tensor Data Mover (wave 0 issues, TENSORcnt-waited, barrier-published).
// Each wave keeps its 16x256 f32 accumulator + Q fragments in VGPRs and runs
// online softmax; P is transposed C->A layout through LDS.
// grid = 256, block = 128 (4 waves).
// ---------------------------------------------------------------------------
__global__ __launch_bounds__(128) void attn_kernel(
    const _Float16* __restrict__ Qh, const _Float16* __restrict__ Kh,
    const _Float16* __restrict__ Vt, const int* __restrict__ mask,
    const float* __restrict__ gamma, const float* __restrict__ beta,
    float* __restrict__ out) {
  __shared__ _Float16 lds_k[2][32][HIDDEN];   // 2 x 16KB, rows = keys
  __shared__ _Float16 lds_v[2][HIDDEN][32];   // 2 x 16KB, rows = hidden
  __shared__ _Float16 lds_p[4][16][32];       // per-wave P transpose staging

  const int lane  = threadIdx.x & 31;
  const int w     = threadIdx.x >> 5;
  const int half_ = lane >> 4;
  const int l16   = lane & 15;

  const int qt = blockIdx.x * 4 + w;   // q tile id in [0,1024)
  const int b  = qt >> 7;              // batch (same for all 4 waves)
  const int q0 = (qt & 127) << 4;      // row within batch
  const size_t qrow = (size_t)qt * 16; // global row base

  const _Float16* Qb = Qh + qrow * HIDDEN;
  const _Float16* Kb = Kh + (size_t)b * S_LEN * HIDDEN;
  const _Float16* Vb = Vt + (size_t)b * HIDDEN * S_LEN;
  const int*      Mb = mask + ((size_t)b * S_LEN + q0) * S_LEN;

  // Cache Q A-fragments for all 8 K-chunks of the hidden dim.
  v16h qa[8];
#pragma unroll
  for (int c = 0; c < 8; ++c) {
    const _Float16* p = Qb + (size_t)l16 * HIDDEN + 32 * c + 16 * half_;
    v8h lo = *(const v8h*)p;
    v8h hi = *(const v8h*)(p + 8);
#pragma unroll
    for (int i = 0; i < 8; ++i) { qa[c][i] = lo[i]; qa[c][i + 8] = hi[i]; }
  }

  const v8f zero8 = {0.f, 0.f, 0.f, 0.f, 0.f, 0.f, 0.f, 0.f};
  v8f acc[16];
#pragma unroll
  for (int t = 0; t < 16; ++t) acc[t] = zero8;

  float mstate[8], lstate[8];
#pragma unroll
  for (int r = 0; r < 8; ++r) { mstate[r] = -3.0e38f; lstate[r] = 0.f; }

  float gam[16], bet[16];
#pragma unroll
  for (int t = 0; t < 16; ++t) {
    gam[t] = gamma[16 * t + l16];
    bet[t] = beta[16 * t + l16];
  }

  // --- TDM: issue K[32 x 256] and V[256 x 32] tile loads into LDS buffer ---
  // 6-arg toolchain form: (g0, g1, g2, g3, extra, cpol); unused groups zero.
  auto issue_tiles = [&](int k0, int buf) {
    const unsigned kdst = (unsigned)(size_t)&lds_k[buf][0][0];
    const unsigned vdst = (unsigned)(size_t)&lds_v[buf][0][0];
    const unsigned long long kga =
        (unsigned long long)(size_t)(Kb + (size_t)k0 * HIDDEN);
    const unsigned long long vga =
        (unsigned long long)(size_t)(Vb + k0);
    const i32x4 gz4 = {0, 0, 0, 0};
    const i32x8 gz8 = {0, 0, 0, 0, 0, 0, 0, 0};
    // K tile: 32 rows x 256 elems, row stride 256 elements (contiguous 16KB)
    __builtin_amdgcn_tensor_load_to_lds(
        tdm_g0(kdst, kga),
        tdm_g1_2d(/*td0=*/HIDDEN, /*td1=*/S_LEN, /*tile0=*/HIDDEN,
                  /*tile1=*/32, /*stride0=*/HIDDEN),
        gz4, gz4, gz8, 0);
    // V tile: 256 rows x 32 elems, row stride S elements
    __builtin_amdgcn_tensor_load_to_lds(
        tdm_g0(vdst, vga),
        tdm_g1_2d(/*td0=*/S_LEN, /*td1=*/HIDDEN, /*tile0=*/32,
                  /*tile1=*/HIDDEN, /*stride0=*/S_LEN),
        gz4, gz4, gz8, 0);
  };

  if (w == 0) issue_tiles(0, 0);

#pragma unroll 1
  for (int j = 0; j < NCHUNK; ++j) {
    const int k0  = j * 32;
    const int buf = j & 1;

    if (w == 0) {
      if (j + 1 < NCHUNK) {
        issue_tiles(k0 + 32, buf ^ 1);
        __builtin_amdgcn_s_wait_tensorcnt(2);  // current buffer's 2 loads done
      } else {
        __builtin_amdgcn_s_wait_tensorcnt(0);
      }
    }
    __syncthreads();  // publish staged K/V to all waves

    // ---- scores: two 16x16 C tiles over keys [k0,k0+32), from LDS ---------
    v8f c0 = zero8, c1 = zero8;
#pragma unroll
    for (int c = 0; c < 8; ++c) {
      v16h kb0 = *(const v16h*)&lds_k[buf][l16][32 * c + 16 * half_];
      v16h kb1 = *(const v16h*)&lds_k[buf][16 + l16][32 * c + 16 * half_];
      c0 = wmma_f16(qa[c], kb0, c0);
      c1 = wmma_f16(qa[c], kb1, c1);
    }
    if (j + 1 < NCHUNK)
      __builtin_prefetch(Mb + (size_t)(8 * half_) * S_LEN + k0 + 32, 0, 0);

    // ---- mask + scale (mask streamed non-temporally: single use) ----------
    float s0[8], s1[8];
#pragma unroll
    for (int r = 0; r < 8; ++r) {
      const size_t mrow = (size_t)(r + 8 * half_) * S_LEN;
      const int m0 = __builtin_nontemporal_load(Mb + mrow + k0 + l16);
      const int m1 = __builtin_nontemporal_load(Mb + mrow + k0 + 16 + l16);
      s0[r] = m0 ? c0[r] * SCALE : NEGBIG;
      s1[r] = m1 ? c1[r] * SCALE : NEGBIG;
    }

    // ---- online softmax ----------------------------------------------------
    float corr[8];
#pragma unroll
    for (int r = 0; r < 8; ++r) {
      const float cmax = rmax16(fmaxf(s0[r], s1[r]));
      const float mnew = fmaxf(mstate[r], cmax);
      const float p0 = __expf(s0[r] - mnew);
      const float p1 = __expf(s1[r] - mnew);
      const float rs = rsum16(p0 + p1);
      corr[r]   = __expf(mstate[r] - mnew);
      lstate[r] = lstate[r] * corr[r] + rs;
      mstate[r] = mnew;
      lds_p[w][r + 8 * half_][l16]      = (_Float16)p0;
      lds_p[w][r + 8 * half_][16 + l16] = (_Float16)p1;
    }
#pragma unroll
    for (int t = 0; t < 16; ++t)
#pragma unroll
      for (int r = 0; r < 8; ++r) acc[t][r] *= corr[r];

    __syncthreads();  // P staging visible (and all waves done with scores)

    // ---- P (C layout) -> A fragment via LDS --------------------------------
    v16h pa;
    {
      const _Float16* pp = &lds_p[w][l16][16 * half_];
      v8h lo = *(const v8h*)pp;
      v8h hi = *(const v8h*)(pp + 8);
#pragma unroll
      for (int i = 0; i < 8; ++i) { pa[i] = lo[i]; pa[i + 8] = hi[i]; }
    }

    // ---- out += P @ V (V fragments from LDS) -------------------------------
#pragma unroll
    for (int t = 0; t < 16; ++t) {
      v16h vbf = *(const v16h*)&lds_v[buf][16 * t + l16][16 * half_];
      acc[t] = wmma_f16(pa, vbf, acc[t]);
    }

    __syncthreads();  // all waves done reading buf before TDM refills it
  }

  // ---- epilogue: 1/l, LayerNorm over H=256, store --------------------------
#pragma unroll
  for (int r = 0; r < 8; ++r) {
    const float inv = 1.0f / lstate[r];
    float psum = 0.f;
#pragma unroll
    for (int t = 0; t < 16; ++t) {
      const float x = acc[t][r] * inv;
      acc[t][r] = x;
      psum += x;
    }
    const float mean = rsum16(psum) * (1.0f / HIDDEN);
    float pvar = 0.f;
#pragma unroll
    for (int t = 0; t < 16; ++t) {
      const float d = acc[t][r] - mean;
      pvar += d * d;
    }
    const float rstd = rsqrtf(rsum16(pvar) * (1.0f / HIDDEN) + 1e-6f);
    const size_t obase = (qrow + r + 8 * half_) * HIDDEN;
#pragma unroll
    for (int t = 0; t < 16; ++t) {
      const float y = (acc[t][r] - mean) * rstd * gam[t] + bet[t];
      __builtin_nontemporal_store(y, out + obase + 16 * t + l16);
    }
  }
}

// ---------------------------------------------------------------------------
extern "C" void kernel_launch(void* const* d_in, const int* in_sizes, int n_in,
                              void* d_out, int out_size, void* d_ws, size_t ws_size,
                              hipStream_t stream) {
  const float* q     = (const float*)d_in[0];
  const float* k     = (const float*)d_in[1];
  const float* v     = (const float*)d_in[2];
  const int*   mask  = (const int*)d_in[3];
  const float* Wq    = (const float*)d_in[4];
  const float* bq    = (const float*)d_in[5];
  const float* Wk    = (const float*)d_in[6];
  const float* bk    = (const float*)d_in[7];
  const float* Wv    = (const float*)d_in[8];
  const float* bv    = (const float*)d_in[9];
  const float* gamma = (const float*)d_in[10];
  const float* beta  = (const float*)d_in[11];

  const size_t n_elems = (size_t)NBATCH * S_LEN * HIDDEN;  // 4M
  _Float16* Qh = (_Float16*)d_ws;
  _Float16* Kh = Qh + n_elems;
  _Float16* Vt = Kh + n_elems;

  proj_kernel<<<dim3(1024, 4, 3), 128, 0, stream>>>(q, k, v, Wq, Wk, Wv,
                                                    bq, bk, bv, Qh, Kh, Vt);
  attn_kernel<<<dim3(256), 128, 0, stream>>>(Qh, Kh, Vt, mask, gamma, beta,
                                             (float*)d_out);
}
